// LowRankLeakyNoisyRNN_91036126806941
// MI455X (gfx1250) — 8-bit matrix operand order repaired and checked by reading, compiled, not run on hardware
//
#include <hip/hip_runtime.h>

#define B_  64
#define T_  2048
#define I_  3
#define H_  512
#define ALPHA     0.2f
#define NOISE_STD 0.05f

typedef __attribute__((ext_vector_type(16))) int   v16i;
typedef __attribute__((ext_vector_type(8)))  float v8f;

#define W_BASE   0
#define W_SIZE   (H_ * H_)
#define A_BASE   (W_BASE + W_SIZE)
#define A_SIZE   (B_ * H_)
#define IN_BASE  (A_BASE + A_SIZE)
#define IN_SIZE  (B_ * I_ * 4)
#define OUT_BASE (IN_BASE + IN_SIZE)
#define OUT_SIZE (B_ * 4)
#define SMEM_BYTES (OUT_BASE + OUT_SIZE)

__device__ __forceinline__ unsigned f32_to_e4m3(float x) {
  unsigned u   = __float_as_uint(x);
  unsigned sgn = (u >> 24) & 0x80u;
  unsigned au  = u & 0x7fffffffu;
  float ax     = __uint_as_float(au);
  if (ax < 0.015625f) {
    int q = (int)rintf(ax * 512.0f);
    return sgn | (unsigned)q;
  }
  unsigned m = au + 0x7ffffu + ((au >> 20) & 1u);
  int e = (int)(m >> 23) - 120;
  if (e >= 16 || !(ax < 448.0f)) return sgn | 0x7eu;
  return sgn | ((unsigned)e << 3) | ((m >> 20) & 7u);
}

__device__ __forceinline__ int a_elem_addr(int row, int k) {
  int v    = ((k >> 6) << 3) + (((k >> 4) & 3) << 1) + ((k >> 2) & 1);
  int lane = (((k >> 3) & 1) << 4) + row;
  return ((v >> 2) << 9) + (lane << 4) + ((v & 3) << 2) + (k & 3);
}
__device__ __forceinline__ int b_elem_addr(int n16, int k) {
  return ((k >> 5) << 9) + (((k >> 3) & 1) << 8) + (n16 << 4) + (((k >> 4) & 1) << 3) + (k & 7);
}
__device__ __forceinline__ v16i load_tile(const char* p, int lane) {
  v16i r;
#pragma unroll
  for (int g = 0; g < 4; ++g) {
    int4 q = *(const int4*)(p + g * 512 + lane * 16);
    r[4 * g + 0] = q.x; r[4 * g + 1] = q.y;
    r[4 * g + 2] = q.z; r[4 * g + 3] = q.w;
  }
  return r;
}

extern "C" __global__ __launch_bounds__(512, 1)
void rnn_persistent(const float* __restrict__ input, const float* __restrict__ noise,
                    const float* __restrict__ w_i,   const float* __restrict__ s_i,
                    const float* __restrict__ m_rec, const float* __restrict__ n_rec,
                    const float* __restrict__ rec_noise,
                    const float* __restrict__ w_o,   const float* __restrict__ s_o,
                    const float* __restrict__ h0,    float* __restrict__ out) {
  extern __shared__ char lds[];
  char*  Wl   = lds + W_BASE;
  char*  Al   = lds + A_BASE;
  float* inl  = (float*)(lds + IN_BASE);
  float* outs = (float*)(lds + OUT_BASE);

  const int tid  = threadIdx.x;
  const int wave = tid >> 5;
  const int lane = tid & 31;
  const int ln   = lane & 15;
  const int hi   = lane >> 4;
  const int mt   = wave >> 2;
  const int ntb  = (wave & 3) * 8;

  const float invH = 1.0f / (float)H_;
  for (int pos = tid; pos < H_ * H_; pos += 512) {
    int n = pos >> 9, k = pos & (H_ - 1);
    float w = m_rec[n] * n_rec[k] * invH + rec_noise[pos];
    int tile = ((n >> 4) << 2) + (k >> 7);
    Wl[tile * 2048 + b_elem_addr(n & 15, k & 127)] = (char)f32_to_e4m3(w * 64.0f);
  }
  for (int pos = tid; pos < B_ * H_; pos += 512) {
    int b = pos >> 9, k = pos & (H_ - 1);
    float r = tanhf(h0[k]);
    int tile = ((b >> 4) << 2) + (k >> 7);
    Al[tile * 2048 + a_elem_addr(b & 15, k & 127)] = (char)f32_to_e4m3(r * 16.0f);
  }
  if (tid < B_) outs[tid] = 0.0f;

  float wi0[8], wi1[8], wi2[8], wo[8], h[64];
#pragma unroll
  for (int j = 0; j < 8; ++j) {
    int n  = (ntb + j) * 16 + ln;
    wi0[j] = w_i[0 * H_ + n] * s_i[0];
    wi1[j] = w_i[1 * H_ + n] * s_i[1];
    wi2[j] = w_i[2 * H_ + n] * s_i[2];
    wo[j]  = w_o[n] * s_o[0] * invH;
    float hn = h0[n];
#pragma unroll
    for (int e = 0; e < 8; ++e) h[j * 8 + e] = hn;
  }

  const v8f vzero = {0.f, 0.f, 0.f, 0.f, 0.f, 0.f, 0.f, 0.f};

  for (int t = 0; t < T_; ++t) {
    if (tid < B_ * I_) {
      int b = tid / 3, i = tid - b * 3;
      inl[tid] = input[(b * T_ + t) * I_ + i];
    }
    __syncthreads();
    if (tid < B_) {
      if (t > 0) out[tid * T_ + (t - 1)] = outs[tid];
      outs[tid] = 0.0f;
    }
    if (t + 1 < T_) {
#pragma unroll
      for (int q = 0; q < 4; ++q) {
        int p = tid * 4 + q;
        __builtin_prefetch((const void*)(noise + ((size_t)(p >> 5) * T_ + (t + 1)) * H_
                                               + (size_t)(p & 31) * 16), 0, 0);
      }
    }

#pragma unroll
    for (int grp = 0; grp < 2; ++grp) {
      v8f acc[4];
#pragma unroll
      for (int j = 0; j < 4; ++j) acc[j] = vzero;
#pragma unroll
      for (int kc = 0; kc < 4; ++kc) {
        v16i Aop  = load_tile(Al + (mt * 4 + kc) * 2048, lane);
        v16i Bcur = load_tile(Wl + ((ntb + grp * 4) * 4 + kc) * 2048, lane);
#pragma unroll
        for (int j = 0; j < 4; ++j) {
          v16i Bnxt = (j < 3)
              ? load_tile(Wl + ((ntb + grp * 4 + j + 1) * 4 + kc) * 2048, lane)
              : Bcur;
          acc[j] = __builtin_amdgcn_wmma_f32_16x16x128_fp8_fp8(
                       Aop, Bcur, (short)0, acc[j], false, false);
          Bcur = Bnxt;
        }
      }
#pragma unroll
      for (int e = 0; e < 8; ++e) {
        const int m = mt * 16 + e + 8 * hi;
        float i0 = inl[m * 3 + 0], i1 = inl[m * 3 + 1], i2 = inl[m * 3 + 2];
#pragma unroll
        for (int j = 0; j < 4; ++j) {
          const int tile = grp * 4 + j;
          const int n    = (ntb + tile) * 16 + ln;
          float nz = noise[((size_t)m * T_ + t) * H_ + n];
          float d  = NOISE_STD * nz
                   + ALPHA * (i0 * wi0[tile] + i1 * wi1[tile] + i2 * wi2[tile]);
          float hv = h[tile * 8 + e];
          h[tile * 8 + e] = hv + d + ALPHA * (acc[j][e] * (1.0f / 1024.0f) - hv);
        }
      }
    }
    __syncthreads();
    float op[8];
#pragma unroll
    for (int e = 0; e < 8; ++e) op[e] = 0.0f;
#pragma unroll
    for (int tile = 0; tile < 8; ++tile) {
      int   n    = (ntb + tile) * 16 + ln;
      char* base = Al + (mt * 4 + (n >> 7)) * 2048;
      int   k127 = n & 127;
#pragma unroll
      for (int e = 0; e < 8; ++e) {
        float r = tanhf(h[tile * 8 + e]);
        base[a_elem_addr(e + 8 * hi, k127)] = (char)f32_to_e4m3(r * 16.0f);
        op[e] += r * wo[tile];
      }
    }
#pragma unroll
    for (int e = 0; e < 8; ++e) {
      float v = op[e];
      v += __shfl_xor(v, 1); v += __shfl_xor(v, 2);
      v += __shfl_xor(v, 4); v += __shfl_xor(v, 8);
      if (ln == 0) atomicAdd(&outs[mt * 16 + e + 8 * hi], v);
    }
  }
  __syncthreads();
  if (tid < B_) out[tid * T_ + (T_ - 1)] = outs[tid];
}

extern "C" void kernel_launch(void* const* d_in, const int* in_sizes, int n_in,
                              void* d_out, int out_size, void* d_ws, size_t ws_size,
                              hipStream_t stream) {
  (void)in_sizes; (void)n_in; (void)out_size; (void)d_ws; (void)ws_size;
  const float* input     = (const float*)d_in[0];
  const float* noise     = (const float*)d_in[1];
  const float* w_i       = (const float*)d_in[2];
  const float* s_i       = (const float*)d_in[3];
  const float* m_rec     = (const float*)d_in[4];
  const float* n_rec     = (const float*)d_in[5];
  const float* rec_noise = (const float*)d_in[6];
  const float* w_o       = (const float*)d_in[7];
  const float* s_o       = (const float*)d_in[8];
  const float* h0        = (const float*)d_in[9];
  float*       out       = (float*)d_out;

  hipFuncSetAttribute(reinterpret_cast<const void*>(rnn_persistent),
                      hipFuncAttributeMaxDynamicSharedMemorySize, SMEM_BYTES);
  hipLaunchKernelGGL(rnn_persistent, dim3(1), dim3(512), SMEM_BYTES, stream,
                     input, noise, w_i, s_i, m_rec, n_rec, rec_noise, w_o, s_o, h0, out);
}
